// CGDN_30932354466053
// MI455X (gfx1250) — matrix operand order repaired and checked by reading, not compiled, not run on hardware
//
#include <hip/hip_runtime.h>
#include <hip/hip_bf16.h>
#include <math.h>

// ---------------------------------------------------------------------------
// Problem constants (match reference)
// ---------------------------------------------------------------------------
#define HID   128
#define HEADS 4
#define CH    32
#define EDIM  4
#define NBLK  4
#define NEG_SLOPE 0.2f
#define LN_EPS 1e-5f
#define MAX_DISP 50.0f

typedef __bf16 bf16_t;
typedef __bf16 v8bf  __attribute__((ext_vector_type(8)));
typedef __bf16 v16bf __attribute__((ext_vector_type(16)));
typedef float  v8f   __attribute__((ext_vector_type(8)));

__device__ __forceinline__ float gelu_exact(float t) {
    return 0.5f * t * (1.0f + erff(t * 0.70710678118654752f));
}
// order-preserving float<->int encoding for atomic max on floats
__device__ __forceinline__ int f2o(float f) {
    int i = __float_as_int(f);
    return (i >= 0) ? i : (i ^ 0x7FFFFFFF);
}
__device__ __forceinline__ float o2f(int i) {
    return __int_as_float((i >= 0) ? i : (i ^ 0x7FFFFFFF));
}

// ---------------------------------------------------------------------------
// generic fill
// ---------------------------------------------------------------------------
__global__ void fill_u32_kernel(unsigned int* p, unsigned int v, long n) {
    long t = (long)blockIdx.x * blockDim.x + threadIdx.x;
    if (t < n) p[t] = v;
}

// ---------------------------------------------------------------------------
// degree count + edge-attr segment sum (for ea_mean of self loops)
// ---------------------------------------------------------------------------
__global__ void deg_kernel(const long long* __restrict__ dst,
                           const float* __restrict__ ea,
                           float* __restrict__ cnt, float* __restrict__ easum,
                           long E_) {
    long e = (long)blockIdx.x * blockDim.x + threadIdx.x;
    if (e >= E_) return;
    long d = dst[e];
    atomicAdd(&cnt[d], 1.0f);
    for (int q = 0; q < EDIM; ++q)
        atomicAdd(&easum[d * EDIM + q], ea[e * EDIM + q]);
}

__global__ void eamean_kernel(float* __restrict__ easum, const float* __restrict__ cnt, long N_) {
    long t = (long)blockIdx.x * blockDim.x + threadIdx.x;
    if (t >= N_ * EDIM) return;
    float c = cnt[t / EDIM];
    easum[t] = easum[t] / fmaxf(c, 1.0f);   // in-place -> ea_mean
}

// ---------------------------------------------------------------------------
// Repack Wl/Wr (f32, [blk][k][n]) -> bf16 WMMA B-operand layout:
//   wpack[blk][w][nt][kt][lane][j], lane<16: col=lane, kk=kt*32+j
//                                   lane>=16: col=lane-16, kk=kt*32+16+j
// ---------------------------------------------------------------------------
__global__ void repack_w_kernel(const float* __restrict__ Wl,
                                const float* __restrict__ Wr,
                                bf16_t* __restrict__ wpack) {
    int t = blockIdx.x * blockDim.x + threadIdx.x;
    if (t >= NBLK * 2 * 8 * 4 * 32 * 16) return;
    int j    =  t        & 15;
    int lane = (t >> 4)  & 31;
    int kt   = (t >> 9)  & 3;
    int nt   = (t >> 11) & 7;
    int w    = (t >> 14) & 1;
    int blk  =  t >> 15;
    int nn = lane & 15;
    int kk = kt * 32 + ((lane >> 4) & 1) * 16 + j;
    int n  = nt * 16 + nn;
    const float* W = w ? Wr : Wl;
    wpack[t] = (bf16_t)W[((long)blk * HID + kk) * HID + n];
}

// ---------------------------------------------------------------------------
// Encoder: h = gelu(LN(x @ enc_W + b)) ; wave32 per node, 4 feats/lane
// ---------------------------------------------------------------------------
__global__ void encoder_kernel(const float* __restrict__ x,
                               const float* __restrict__ encW,
                               const float* __restrict__ encb,
                               const float* __restrict__ lng,
                               const float* __restrict__ lnb,
                               float* __restrict__ h, bf16_t* __restrict__ hb, long N_) {
    __shared__ float s[6 * HID + 3 * HID];
    int tid = threadIdx.x;
    for (int i = tid; i < 6 * HID; i += blockDim.x) s[i] = encW[i];
    for (int i = tid; i < HID; i += blockDim.x) {
        s[768 + i]  = encb[i];
        s[896 + i]  = lng[i];
        s[1024 + i] = lnb[i];
    }
    __syncthreads();
    int wave = tid >> 5, lane = tid & 31;
    long node = (long)blockIdx.x * 8 + wave;
    if (node >= N_) return;
    float xv[6];
    for (int q = 0; q < 6; ++q) xv[q] = x[node * 6 + q];
    float y[4], sum = 0.0f;
    for (int u = 0; u < 4; ++u) {
        int j = lane * 4 + u;
        float a = s[768 + j];
        for (int q = 0; q < 6; ++q) a += xv[q] * s[q * HID + j];
        y[u] = a; sum += a;
    }
    for (int o = 16; o; o >>= 1) sum += __shfl_xor(sum, o, 32);
    float mu = sum * (1.0f / HID), v2 = 0.0f;
    for (int u = 0; u < 4; ++u) { float d = y[u] - mu; v2 += d * d; }
    for (int o = 16; o; o >>= 1) v2 += __shfl_xor(v2, o, 32);
    float rstd = rsqrtf(v2 * (1.0f / HID) + LN_EPS);
    for (int u = 0; u < 4; ++u) {
        int j = lane * 4 + u;
        float t = (y[u] - mu) * rstd * s[896 + j] + s[1024 + j];
        float g = gelu_exact(t);
        h[node * HID + j]  = g;
        hb[node * HID + j] = (bf16_t)g;
    }
}

// ---------------------------------------------------------------------------
// FiLM: gamma/beta from target_mp (tiny, one block)
// ---------------------------------------------------------------------------
__global__ void film_kernel(const float* __restrict__ tmp_,
                            const float* __restrict__ W1, const float* __restrict__ b1,
                            const float* __restrict__ W2, const float* __restrict__ b2,
                            float* __restrict__ gb) {
    __shared__ float a[64];
    int t = threadIdx.x;
    float tv = tmp_[0];
    if (t < 64) a[t] = gelu_exact(tv * W1[t] + b1[t]);
    __syncthreads();
    if (t < 2 * HID) {
        float acc = b2[t];
        for (int i = 0; i < 64; ++i) acc += a[i] * W2[i * (2 * HID) + t];
        gb[t] = acc;
    }
}

// ---------------------------------------------------------------------------
// WMMA GEMM: xl = h@Wl + bl, xr = h@Wr + br  (bf16 in, f32 acc)
// One 256-thread WG: stage both packed weight matrices (64KB) in LDS,
// each wave computes a 16x128 output tile with 64 v_wmma_f32_16x16x32_bf16.
// ---------------------------------------------------------------------------
__global__ __launch_bounds__(256)
void gemm_xlxr_kernel(const bf16_t* __restrict__ hb,
                      const bf16_t* __restrict__ wpack,   // [2][8][4][32][16] bf16
                      const float* __restrict__ bl, const float* __restrict__ br,
                      float* __restrict__ xl, float* __restrict__ xr, long nrows) {
    __shared__ __align__(32) bf16_t wlds[2 * 8 * 4 * 32 * 16];  // 64 KB
    const int tid = threadIdx.x;
    // cooperative 16B-vector staging of weights into LDS
    for (int i = tid; i < 4096; i += 256)
        ((uint4*)wlds)[i] = ((const uint4*)wpack)[i];
    __syncthreads();

    const int wave = tid >> 5;
    const int lane = tid & 31;
    const long m0 = ((long)blockIdx.x * 8 + wave) * 16;
    if (m0 >= nrows) return;                 // uniform per wave; full tiles only (N % 16 == 0)

    const int laneHi = (lane >> 4) & 1;      // K-half select (contiguous k order, same as wpack B)
    const int lrow   = lane & 15;            // A row within tile
    const bf16_t* arow = hb + (m0 + lrow) * HID;

    for (int w = 0; w < 2; ++w) {
        v8f acc[8] = {};
        for (int kt = 0; kt < 4; ++kt) {
            const int kbase = kt * 32 + laneHi * 16;
            // A fragment: lane<16 -> K = kt*32+{0..15}; lane>=16 -> +16
            v8bf alo = *(const v8bf*)(arow + kbase);
            v8bf ahi = *(const v8bf*)(arow + kbase + 8);
            v16bf afrag;
            for (int i = 0; i < 8; ++i) { afrag[i] = alo[i]; afrag[i + 8] = ahi[i]; }
            for (int nt = 0; nt < 8; ++nt) {
                const bf16_t* bp = wlds + (((w * 8 + nt) * 4 + kt) * 32 + lane) * 16;
                v16bf bfrag = *(const v16bf*)bp;     // 32B aligned LDS read
                acc[nt] = __builtin_amdgcn_wmma_f32_16x16x32_bf16(
                    false, afrag, false, bfrag, (short)0, acc[nt], false, false);
            }
        }
        const float* bias = w ? br : bl;
        float* out = w ? xr : xl;
        for (int nt = 0; nt < 8; ++nt) {
            const int n  = nt * 16 + lrow;       // C/D layout: col = lane&15
            const float bn = bias[n];
            for (int r = 0; r < 8; ++r) {        // row = r + 8*laneHi
                long m = m0 + r + laneHi * 8;
                out[m * HID + n] = acc[nt][r] + bn;
            }
        }
    }
}

// ---------------------------------------------------------------------------
// Edge pass A: attention logits + segment max (wave32 per edge)
// ---------------------------------------------------------------------------
__global__ void edge_logits_kernel(const long long* __restrict__ src,
                                   const long long* __restrict__ dst,
                                   const float* __restrict__ ea,
                                   const float* __restrict__ eamean,
                                   const float* __restrict__ We,   // [4][128] this block
                                   const float* __restrict__ att,  // [H][C]  this block
                                   const float* __restrict__ xl,
                                   const float* __restrict__ xr,
                                   float* __restrict__ alpha, int* __restrict__ amax,
                                   long E_, long N_) {
    __shared__ float sWe[EDIM * HID];
    __shared__ float sAtt[HID];
    int tid = threadIdx.x;
    for (int i = tid; i < EDIM * HID; i += 256) sWe[i] = We[i];
    for (int i = tid; i < HID; i += 256) sAtt[i] = att[i];
    __syncthreads();
    int wave = tid >> 5, lane = tid & 31;
    long e = (long)blockIdx.x * 8 + wave;
    if (e >= E_ + N_) return;
    long s, d; float eav[EDIM];
    if (e < E_) {
        s = src[e]; d = dst[e];
        for (int q = 0; q < EDIM; ++q) eav[q] = ea[e * EDIM + q];
    } else {
        s = d = e - E_;
        for (int q = 0; q < EDIM; ++q) eav[q] = eamean[(e - E_) * EDIM + q];
    }
    __builtin_prefetch(&xl[s * HID], 0, 0);
    __builtin_prefetch(&xr[d * HID], 0, 0);
    float part = 0.0f;
    for (int u = 0; u < 4; ++u) {
        int j = lane * 4 + u;
        float ee = eav[0] * sWe[j] + eav[1] * sWe[HID + j]
                 + eav[2] * sWe[2 * HID + j] + eav[3] * sWe[3 * HID + j];
        float m = xl[s * HID + j] + xr[d * HID + j] + ee;
        float a = (m > 0.0f) ? m : NEG_SLOPE * m;
        part += a * sAtt[j];             // att flat: j = h*32 + c
    }
    // reduce within 8-lane head group (head = lane>>3)
    part += __shfl_xor(part, 1, 32);
    part += __shfl_xor(part, 2, 32);
    part += __shfl_xor(part, 4, 32);
    if ((lane & 7) == 0) {
        int h = lane >> 3;
        alpha[e * HEADS + h] = part;
        atomicMax(&amax[d * HEADS + h], f2o(part));
    }
}

// ---------------------------------------------------------------------------
// Edge pass B: p = exp(alpha - max), denom accumulation (thread per (e,h))
// ---------------------------------------------------------------------------
__global__ void edge_softmax_kernel(const long long* __restrict__ dst,
                                    float* __restrict__ alpha,   // in logits, out p
                                    const int* __restrict__ amax,
                                    float* __restrict__ denom, long E_, long N_) {
    long t = (long)blockIdx.x * blockDim.x + threadIdx.x;
    if (t >= (E_ + N_) * HEADS) return;
    long e = t >> 2; int h = (int)(t & 3);
    long d = (e < E_) ? dst[e] : (e - E_);
    float p = __expf(alpha[t] - o2f(amax[d * HEADS + h]));
    alpha[t] = p;
    atomicAdd(&denom[d * HEADS + h], p);
}

// ---------------------------------------------------------------------------
// Edge pass C: acc[dst] += (p/denom) * xl[src]  (wave32 per edge, fp32 atomics)
// ---------------------------------------------------------------------------
__global__ void edge_scatter_kernel(const long long* __restrict__ src,
                                    const long long* __restrict__ dst,
                                    const float* __restrict__ p,
                                    const float* __restrict__ denom,
                                    const float* __restrict__ xl,
                                    float* __restrict__ acc, long E_, long N_) {
    int tid = threadIdx.x;
    int wave = tid >> 5, lane = tid & 31;
    long e = (long)blockIdx.x * 8 + wave;
    if (e >= E_ + N_) return;
    long s, d;
    if (e < E_) { s = src[e]; d = dst[e]; } else { s = d = e - E_; }
    int h = lane >> 3;                                    // 4 feats/lane, same head
    float w = p[e * HEADS + h] / denom[d * HEADS + h];
    for (int u = 0; u < 4; ++u) {
        int j = lane * 4 + u;
        atomicAdd(&acc[d * HID + j], w * xl[s * HID + j]);
    }
}

// ---------------------------------------------------------------------------
// Node update: h = gelu(LN(gamma*(acc+bias)+beta)) + h_res ; wave32 per node
// ---------------------------------------------------------------------------
__global__ void node_update_kernel(const float* __restrict__ acc,
                                   const float* __restrict__ bias_out,
                                   const float* __restrict__ gb,   // gamma|beta
                                   const float* __restrict__ lng,
                                   const float* __restrict__ lnb,
                                   float* __restrict__ h, bf16_t* __restrict__ hb, long N_) {
    __shared__ float s[5 * HID];
    int tid = threadIdx.x;
    for (int i = tid; i < HID; i += 256) {
        s[i]            = bias_out[i];
        s[HID + i]      = gb[i];          // gamma
        s[2 * HID + i]  = gb[HID + i];    // beta
        s[3 * HID + i]  = lng[i];
        s[4 * HID + i]  = lnb[i];
    }
    __syncthreads();
    int wave = tid >> 5, lane = tid & 31;
    long node = (long)blockIdx.x * 8 + wave;
    if (node >= N_) return;
    float y[4], sum = 0.0f;
    for (int u = 0; u < 4; ++u) {
        int j = lane * 4 + u;
        float v = acc[node * HID + j] + s[j];
        v = s[HID + j] * v + s[2 * HID + j];
        y[u] = v; sum += v;
    }
    for (int o = 16; o; o >>= 1) sum += __shfl_xor(sum, o, 32);
    float mu = sum * (1.0f / HID), v2 = 0.0f;
    for (int u = 0; u < 4; ++u) { float d = y[u] - mu; v2 += d * d; }
    for (int o = 16; o; o >>= 1) v2 += __shfl_xor(v2, o, 32);
    float rstd = rsqrtf(v2 * (1.0f / HID) + LN_EPS);
    for (int u = 0; u < 4; ++u) {
        int j = lane * 4 + u;
        float t = (y[u] - mu) * rstd * s[3 * HID + j] + s[4 * HID + j];
        float hn = gelu_exact(t) + h[node * HID + j];
        h[node * HID + j]  = hn;
        hb[node * HID + j] = (bf16_t)hn;
    }
}

// ---------------------------------------------------------------------------
// Decoder + output assembly (new_coords then delta, concatenated flat)
// ---------------------------------------------------------------------------
__global__ __launch_bounds__(256)
void decoder_kernel(const float* __restrict__ h, const float* __restrict__ x,
                    const unsigned char* __restrict__ fixed_mask,
                    const float* __restrict__ W1, const float* __restrict__ b1,
                    const float* __restrict__ W2, const float* __restrict__ b2,
                    float* __restrict__ out, long N_) {
    __shared__ float sW1[HID * 64];     // 32 KB
    __shared__ float sb1[64], sW2[128], sb2[2];
    __shared__ float sh[8][HID];
    int tid = threadIdx.x;
    for (int i = tid; i < HID * 64; i += 256) sW1[i] = W1[i];
    for (int i = tid; i < 64; i += 256) sb1[i] = b1[i];
    for (int i = tid; i < 128; i += 256) sW2[i] = W2[i];
    if (tid < 2) sb2[tid] = b2[tid];
    int wave = tid >> 5, lane = tid & 31;
    long node = (long)blockIdx.x * 8 + wave;
    if (node < N_)
        for (int u = 0; u < 4; ++u) sh[wave][lane * 4 + u] = h[node * HID + lane * 4 + u];
    __syncthreads();
    if (node >= N_) return;
    float d0 = 0.0f, d1 = 0.0f;
    for (int ii = 0; ii < 2; ++ii) {
        int i = lane * 2 + ii;
        float a = sb1[i];
        for (int k = 0; k < HID; ++k) a += sh[wave][k] * sW1[k * 64 + i];
        a = gelu_exact(a);
        d0 += a * sW2[i * 2 + 0];
        d1 += a * sW2[i * 2 + 1];
    }
    for (int o = 16; o; o >>= 1) { d0 += __shfl_xor(d0, o, 32); d1 += __shfl_xor(d1, o, 32); }
    if (lane == 0) {
        d0 += sb2[0]; d1 += sb2[1];
        d0 = fminf(fmaxf(d0, -MAX_DISP), MAX_DISP);
        d1 = fminf(fmaxf(d1, -MAX_DISP), MAX_DISP);
        float free_ = fixed_mask[node] ? 0.0f : 1.0f;
        d0 *= free_; d1 *= free_;
        out[node * 2 + 0] = x[node * 6 + 0] + d0;       // new_coords
        out[node * 2 + 1] = x[node * 6 + 1] + d1;
        out[2 * N_ + node * 2 + 0] = d0;                // delta
        out[2 * N_ + node * 2 + 1] = d1;
    }
}

// ---------------------------------------------------------------------------
// Host-side orchestration
// ---------------------------------------------------------------------------
extern "C" void kernel_launch(void* const* d_in, const int* in_sizes, int n_in,
                              void* d_out, int out_size, void* d_ws, size_t ws_size,
                              hipStream_t stream) {
    const float*      x        = (const float*)d_in[0];
    const long long*  eidx     = (const long long*)d_in[1];   // int64 per reference
    const float*      eattr    = (const float*)d_in[2];
    const float*      tmp_     = (const float*)d_in[3];
    const unsigned char* fmask = (const unsigned char*)d_in[4]; // bool
    const float* enc_W   = (const float*)d_in[5];
    const float* enc_b   = (const float*)d_in[6];
    const float* enc_lng = (const float*)d_in[7];
    const float* enc_lnb = (const float*)d_in[8];
    const float* film_W1 = (const float*)d_in[9];
    const float* film_b1 = (const float*)d_in[10];
    const float* film_W2 = (const float*)d_in[11];
    const float* film_b2 = (const float*)d_in[12];
    const float* Wl      = (const float*)d_in[13];
    const float* bl      = (const float*)d_in[14];
    const float* Wr      = (const float*)d_in[15];
    const float* br      = (const float*)d_in[16];
    const float* att     = (const float*)d_in[17];
    const float* We      = (const float*)d_in[18];
    const float* bias_o  = (const float*)d_in[19];
    const float* ln_g    = (const float*)d_in[20];
    const float* ln_b    = (const float*)d_in[21];
    const float* dec_W1  = (const float*)d_in[22];
    const float* dec_b1  = (const float*)d_in[23];
    const float* dec_W2  = (const float*)d_in[24];
    const float* dec_b2  = (const float*)d_in[25];

    const long N_ = in_sizes[0] / 6;
    const long E_ = in_sizes[1] / 2;
    const long EA = E_ + N_;
    const long long* src = eidx;
    const long long* dst = eidx + E_;

    // workspace carve-out
    char* base = (char*)d_ws;
    size_t off = 0;
    auto carve = [&](size_t bytes) { size_t o = off; off = (off + bytes + 255) & ~(size_t)255; return o; };
    float*   h_buf  = (float*)  (base + carve((size_t)N_ * HID * 4));
    bf16_t*  hb_buf = (bf16_t*) (base + carve((size_t)N_ * HID * 2));
    float*   xl_buf = (float*)  (base + carve((size_t)N_ * HID * 4));
    float*   xr_buf = (float*)  (base + carve((size_t)N_ * HID * 4));
    float*   acc_bf = (float*)  (base + carve((size_t)N_ * HID * 4));
    float*   alpha  = (float*)  (base + carve((size_t)EA * HEADS * 4));
    int*     amax   = (int*)    (base + carve((size_t)N_ * HEADS * 4));
    float*   denom  = (float*)  (base + carve((size_t)N_ * HEADS * 4));
    float*   cnt    = (float*)  (base + carve((size_t)N_ * 4));
    float*   easum  = (float*)  (base + carve((size_t)N_ * EDIM * 4));
    bf16_t*  wpack  = (bf16_t*) (base + carve((size_t)NBLK * 2 * HID * HID * 2));
    float*   gb     = (float*)  (base + carve(2 * HID * 4));
    (void)ws_size; (void)n_in; (void)out_size;

    const int TB = 256;
    auto blocks = [](long work, int per) { return (unsigned)((work + per - 1) / per); };

    // --- self-loop edge-attr mean ---
    fill_u32_kernel<<<blocks(N_, TB), TB, 0, stream>>>((unsigned*)cnt, 0u, N_);
    fill_u32_kernel<<<blocks(N_ * EDIM, TB), TB, 0, stream>>>((unsigned*)easum, 0u, N_ * EDIM);
    deg_kernel<<<blocks(E_, TB), TB, 0, stream>>>(dst, eattr, cnt, easum, E_);
    eamean_kernel<<<blocks(N_ * EDIM, TB), TB, 0, stream>>>(easum, cnt, N_);

    // --- weight repack for WMMA ---
    repack_w_kernel<<<blocks(NBLK * 2 * HID * HID, TB), TB, 0, stream>>>(Wl, Wr, wpack);

    // --- encoder + FiLM ---
    encoder_kernel<<<blocks(N_ * 32, TB), TB, 0, stream>>>(x, enc_W, enc_b, enc_lng, enc_lnb,
                                                           h_buf, hb_buf, N_);
    film_kernel<<<1, 256, 0, stream>>>(tmp_, film_W1, film_b1, film_W2, film_b2, gb);

    // --- GATv2 blocks ---
    const long mtiles = N_ / 16;                       // N % 16 == 0 for N=50000
    const unsigned gemm_grid = (unsigned)((mtiles + 7) / 8);
    for (int blk = 0; blk < NBLK; ++blk) {
        fill_u32_kernel<<<blocks(N_ * HID, TB), TB, 0, stream>>>((unsigned*)acc_bf, 0u, N_ * HID);
        fill_u32_kernel<<<blocks(N_ * HEADS, TB), TB, 0, stream>>>((unsigned*)amax, 0x80000000u, N_ * HEADS);
        fill_u32_kernel<<<blocks(N_ * HEADS, TB), TB, 0, stream>>>((unsigned*)denom, 0u, N_ * HEADS);

        gemm_xlxr_kernel<<<gemm_grid, TB, 0, stream>>>(
            hb_buf, wpack + (size_t)blk * 2 * HID * HID,
            bl + (size_t)blk * HID, br + (size_t)blk * HID, xl_buf, xr_buf, N_);

        edge_logits_kernel<<<blocks(EA * 32, TB), TB, 0, stream>>>(
            src, dst, eattr, easum,
            We + (size_t)blk * EDIM * HID, att + (size_t)blk * HEADS * CH,
            xl_buf, xr_buf, alpha, amax, E_, N_);

        edge_softmax_kernel<<<blocks(EA * HEADS, TB), TB, 0, stream>>>(
            dst, alpha, amax, denom, E_, N_);

        edge_scatter_kernel<<<blocks(EA * 32, TB), TB, 0, stream>>>(
            src, dst, alpha, denom, xl_buf, acc_bf, E_, N_);

        node_update_kernel<<<blocks(N_ * 32, TB), TB, 0, stream>>>(
            acc_bf, bias_o + (size_t)blk * HID, gb,
            ln_g + (size_t)blk * HID, ln_b + (size_t)blk * HID, h_buf, hb_buf, N_);
    }

    // --- decoder / outputs ---
    decoder_kernel<<<blocks(N_ * 32, TB), TB, 0, stream>>>(
        h_buf, x, fmask, dec_W1, dec_b1, dec_W2, dec_b2, (float*)d_out, N_);
}
